// TemporalBlock_34935263985742
// MI455X (gfx1250) — matrix operand order repaired and checked by reading, not compiled, not run on hardware
//
#include <hip/hip_runtime.h>
#include <hip/hip_bf16.h>
#include <math.h>

#define DEVI __device__ __forceinline__

typedef __attribute__((ext_vector_type(8)))  __bf16 v8bf;
typedef __attribute__((ext_vector_type(16))) __bf16 v16bf;
typedef __attribute__((ext_vector_type(8)))  float  v8f;

namespace cfg {
constexpr int B    = 8;
constexpr int CIN  = 512;
constexpr int COUT = 512;
constexpr int L    = 4096;
constexpr int KW   = 7;
constexpr int DIL  = 4;
constexpr int PAD  = 24;
constexpr int H    = 2;
constexpr int DH   = 256;          // COUT / H
constexpr int LP   = L + PAD;      // 4120
constexpr int NPOS = LP / DIL;     // 1030
constexpr int G    = B * DIL;      // 32 grouped batches
constexpr int MATT = G * NPOS;     // 32960 attention rows

constexpr int BM = 128, BN = 128, BK = 32, NT = 256;
constexpr int LDK = BK + 8;        // padded LDS leading dim (80B) to break bank conflicts
}

// ---------------- WMMA helpers (wave32, gfx1250) ----------------

DEVI v8f wmma_bf16(v16bf a, v16bf b, v8f c) {
  // v_wmma_f32_16x16x32_bf16: D = A(16x32) x B(32x16) + C(16x16 f32)
  return __builtin_amdgcn_wmma_f32_16x16x32_bf16(
      /*neg_a=*/false, a, /*neg_b=*/false, b,
      /*c_mod=*/(short)0, c, /*reuse_a=*/false, /*reuse_b=*/false);
}

// A tile in LDS: row-major [BM][LDK] bf16.
// ISA 16-bit A 16x32 layout: lanes 0-15 row M=lane, K={0..7,16..23};
// lanes 16-31 row M=lane-16, K={8..15,24..31}.
DEVI v16bf ldsA_frag(const __bf16* s, int row_base) {
  int lane = threadIdx.x & 31;
  const __bf16* p = s + (row_base + (lane & 15)) * cfg::LDK + ((lane & 16) ? 8 : 0);
  v8bf lo = *(const v8bf*)(p);
  v8bf hi = *(const v8bf*)(p + 16);
  return __builtin_shufflevector(lo, hi, 0,1,2,3,4,5,6,7,8,9,10,11,12,13,14,15);
}

// B tile in LDS: column-major [BN][LDK] bf16 (K contiguous per column).
// ISA 16-bit B 32x16 layout: lanes 0-15 col N=lane, K={0..7,16..23};
// lanes 16-31 col N=lane-16, K={8..15,24..31}.
DEVI v16bf ldsB_frag(const __bf16* s, int col_base) {
  int lane = threadIdx.x & 31;
  const __bf16* p = s + (col_base + (lane & 15)) * cfg::LDK + ((lane & 16) ? 8 : 0);
  v8bf lo = *(const v8bf*)(p);
  v8bf hi = *(const v8bf*)(p + 16);
  return __builtin_shufflevector(lo, hi, 0,1,2,3,4,5,6,7,8,9,10,11,12,13,14,15);
}

// ---------------- packing kernels ----------------

__global__ __launch_bounds__(256) void pack_x_k(const float* __restrict__ x,
                                                __bf16* __restrict__ xT) {
  using namespace cfg;
  size_t idx = (size_t)blockIdx.x * blockDim.x + threadIdx.x;
  if (idx >= (size_t)B * CIN * L) return;
  int l = (int)(idx % L);
  size_t r = idx / L;
  int c = (int)(r % CIN);
  int b = (int)(r / CIN);
  xT[((size_t)b * L + l) * CIN + c] = (__bf16)x[idx];
}

// conv1_w (COUT,CIN,KW) -> wconv[k][o][c] (B operand for tap k, [N][K] layout)
__global__ __launch_bounds__(256) void pack_wconv_k(const float* __restrict__ w,
                                                    __bf16* __restrict__ wp) {
  using namespace cfg;
  size_t idx = (size_t)blockIdx.x * blockDim.x + threadIdx.x;
  if (idx >= (size_t)COUT * CIN * KW) return;
  int k = (int)(idx % KW);
  size_t r = idx / KW;
  int c = (int)(r % CIN);
  int o = (int)(r / CIN);
  wp[((size_t)k * COUT + o) * CIN + c] = (__bf16)w[idx];
}

// src [kd][nd] row-major -> dst [nd][kd]  (transpose to [N][K] layout)
__global__ __launch_bounds__(256) void pack_bmat_k(const float* __restrict__ src,
                                                   __bf16* __restrict__ dst,
                                                   int kd, int nd) {
  size_t idx = (size_t)blockIdx.x * blockDim.x + threadIdx.x;
  if (idx >= (size_t)kd * nd) return;
  int j = (int)(idx % nd);
  int c = (int)(idx / nd);
  dst[(size_t)j * kd + c] = (__bf16)src[idx];
}

// down_w (COUT,CIN,1) is already [N][K]; straight convert.
__global__ __launch_bounds__(256) void pack_wdown_k(const float* __restrict__ src,
                                                    __bf16* __restrict__ dst) {
  using namespace cfg;
  size_t idx = (size_t)blockIdx.x * blockDim.x + threadIdx.x;
  if (idx >= (size_t)COUT * CIN) return;
  dst[idx] = (__bf16)src[idx];
}

// Zero the left-padded rows of att_x (lp < PAD)
__global__ __launch_bounds__(256) void zero_edge_k(__bf16* __restrict__ attx) {
  using namespace cfg;
  size_t idx = (size_t)blockIdx.x * blockDim.x + threadIdx.x;
  if (idx >= (size_t)B * PAD * COUT) return;
  int c = (int)(idx % COUT);
  size_t r = idx / COUT;
  int lp = (int)(r % PAD);
  int b = (int)(r / PAD);
  int g = b * DIL + (lp & (DIL - 1));
  int i = lp / DIL;
  attx[((size_t)g * NPOS + i) * COUT + c] = (__bf16)0.0f;
}

// ---------------- conv GEMM: y = relu(dilated_conv(x) + b), scattered to att layout ----------------

__global__ __launch_bounds__(cfg::NT) void conv_gemm_k(const __bf16* __restrict__ xT,
                                                       const __bf16* __restrict__ wconv,
                                                       const float* __restrict__ bias,
                                                       __bf16* __restrict__ attx) {
  using namespace cfg;
  constexpr int TM = L / BM;     // 32
  constexpr int TN = COUT / BN;  // 4
  __shared__ __align__(16) __bf16 sA[BM * LDK];
  __shared__ __align__(16) __bf16 sB[BN * LDK];

  int bid = blockIdx.x;
  int b   = bid / (TM * TN);
  int rr  = bid % (TM * TN);
  int tm  = rr / TN, tn = rr % TN;
  int t0 = tm * BM, n0 = tn * BN;
  int tid = threadIdx.x, lane = tid & 31, wave = tid >> 5;
  int wm = wave >> 1, wn = wave & 1;

  v8f acc[2][4] = {};

  for (int kk = 0; kk < KW; ++kk) {
    for (int c0 = 0; c0 < CIN; c0 += BK) {
      int rowA = tid >> 1;
      int ha   = (tid & 1) * 16;
      // stage A: 128 rows x 32 k, zero-filled when tap falls outside [0,L)
      {
        int tg = t0 + rowA + kk * DIL - PAD;
        uint4 d0 = {0u, 0u, 0u, 0u}, d1 = {0u, 0u, 0u, 0u};
        if (tg >= 0 && tg < L) {
          const __bf16* src = xT + ((size_t)b * L + tg) * CIN + c0 + ha;
          d0 = *(const uint4*)(src);
          d1 = *(const uint4*)(src + 8);
          if (c0 + BK < CIN) __builtin_prefetch(src + BK, 0, 1);
        }
        *(uint4*)(sA + rowA * LDK + ha)     = d0;
        *(uint4*)(sA + rowA * LDK + ha + 8) = d1;
      }
      // stage B: 128 cols x 32 k from wconv[kk]
      {
        const __bf16* src = wconv + ((size_t)kk * COUT + n0 + rowA) * CIN + c0 + ha;
        *(uint4*)(sB + rowA * LDK + ha)     = *(const uint4*)(src);
        *(uint4*)(sB + rowA * LDK + ha + 8) = *(const uint4*)(src + 8);
        if (c0 + BK < CIN) __builtin_prefetch(src + BK, 0, 1);
      }
      __syncthreads();
      v16bf a0 = ldsA_frag(sA, wm * 32);
      v16bf a1 = ldsA_frag(sA, wm * 32 + 16);
      v16bf bf[4];
#pragma unroll
      for (int ni = 0; ni < 4; ++ni) bf[ni] = ldsB_frag(sB, wn * 64 + ni * 16);
#pragma unroll
      for (int ni = 0; ni < 4; ++ni) {
        acc[0][ni] = wmma_bf16(a0, bf[ni], acc[0][ni]);
        acc[1][ni] = wmma_bf16(a1, bf[ni], acc[1][ni]);
      }
      __syncthreads();
    }
  }

  int lr = lane & 15, lhi = (lane >> 4) & 1;
#pragma unroll
  for (int mi = 0; mi < 2; ++mi) {
#pragma unroll
    for (int ni = 0; ni < 4; ++ni) {
      int cg = n0 + wn * 64 + ni * 16 + lr;
      float bv = bias[cg];
#pragma unroll
      for (int i = 0; i < 8; ++i) {
        int t = t0 + wm * 32 + mi * 16 + i + 8 * lhi;   // conv output position
        float v = acc[mi][ni][i] + bv;
        v = fmaxf(v, 0.0f);
        int lp = t + PAD;                 // padded position >= PAD
        int d  = lp & (DIL - 1);
        int ii = lp / DIL;
        int g  = b * DIL + d;
        attx[((size_t)g * NPOS + ii) * COUT + cg] = (__bf16)v;
      }
    }
  }
}

// ---------------- qkv GEMM: qkv = att_x @ qkv_w + b, q scaled by Dh^-0.5 ----------------

__global__ __launch_bounds__(cfg::NT) void qkv_gemm_k(const __bf16* __restrict__ A,
                                                      const __bf16* __restrict__ Bw,
                                                      const float* __restrict__ bias,
                                                      __bf16* __restrict__ out) {
  using namespace cfg;
  constexpr int ND = 3 * COUT;                    // 1536
  constexpr int KD = COUT;                        // 512
  constexpr int TN = ND / BN;                     // 12
  __shared__ __align__(16) __bf16 sA[BM * LDK];
  __shared__ __align__(16) __bf16 sB[BN * LDK];

  int bid = blockIdx.x;
  int tm = bid / TN, tn = bid % TN;
  int m0 = tm * BM, n0 = tn * BN;
  int tid = threadIdx.x, lane = tid & 31, wave = tid >> 5;
  int wm = wave >> 1, wn = wave & 1;

  v8f acc[2][4] = {};

  for (int c0 = 0; c0 < KD; c0 += BK) {
    int rowA = tid >> 1;
    int ha   = (tid & 1) * 16;
    {
      int m = m0 + rowA;
      uint4 d0 = {0u, 0u, 0u, 0u}, d1 = {0u, 0u, 0u, 0u};
      if (m < MATT) {
        const __bf16* src = A + (size_t)m * KD + c0 + ha;
        d0 = *(const uint4*)(src);
        d1 = *(const uint4*)(src + 8);
        if (c0 + BK < KD) __builtin_prefetch(src + BK, 0, 1);
      }
      *(uint4*)(sA + rowA * LDK + ha)     = d0;
      *(uint4*)(sA + rowA * LDK + ha + 8) = d1;
    }
    {
      const __bf16* src = Bw + (size_t)(n0 + rowA) * KD + c0 + ha;
      *(uint4*)(sB + rowA * LDK + ha)     = *(const uint4*)(src);
      *(uint4*)(sB + rowA * LDK + ha + 8) = *(const uint4*)(src + 8);
      if (c0 + BK < KD) __builtin_prefetch(src + BK, 0, 1);
    }
    __syncthreads();
    v16bf a0 = ldsA_frag(sA, wm * 32);
    v16bf a1 = ldsA_frag(sA, wm * 32 + 16);
    v16bf bf[4];
#pragma unroll
    for (int ni = 0; ni < 4; ++ni) bf[ni] = ldsB_frag(sB, wn * 64 + ni * 16);
#pragma unroll
    for (int ni = 0; ni < 4; ++ni) {
      acc[0][ni] = wmma_bf16(a0, bf[ni], acc[0][ni]);
      acc[1][ni] = wmma_bf16(a1, bf[ni], acc[1][ni]);
    }
    __syncthreads();
  }

  int lr = lane & 15, lhi = (lane >> 4) & 1;
#pragma unroll
  for (int mi = 0; mi < 2; ++mi) {
#pragma unroll
    for (int ni = 0; ni < 4; ++ni) {
      int cg = n0 + wn * 64 + ni * 16 + lr;
      float bv = bias[cg];
      float scale = (cg < COUT) ? 0.0625f : 1.0f;   // Dh^-0.5 = 1/16 on Q
#pragma unroll
      for (int i = 0; i < 8; ++i) {
        int m = m0 + wm * 32 + mi * 16 + i + 8 * lhi;
        if (m < MATT) {
          float v = (acc[mi][ni][i] + bv) * scale;
          out[(size_t)m * ND + cg] = (__bf16)v;
        }
      }
    }
  }
}

// ---------------- neighborhood attention (1 wave per (g,i,h)) ----------------

__global__ __launch_bounds__(256) void na1d_attn_k(const __bf16* __restrict__ qkv,
                                                   const float* __restrict__ rpb,
                                                   __bf16* __restrict__ attout) {
  using namespace cfg;
  int wid = blockIdx.x * 8 + (threadIdx.x >> 5);
  if (wid >= G * NPOS * H) return;
  int lane = threadIdx.x & 31;
  int h = wid % H;
  int i = (wid / H) % NPOS;
  int g = wid / (H * NPOS);

  size_t row = (size_t)g * NPOS + i;
  const __bf16* qp = qkv + row * (3 * COUT) + h * DH + lane * 8;

  int start = i - KW / 2;
  if (start < 0) start = 0;
  if (start > NPOS - KW) start = NPOS - KW;

  float s[KW];
#pragma unroll
  for (int j = 0; j < KW; ++j) {
    const __bf16* kp = qkv + ((size_t)g * NPOS + start + j) * (3 * COUT)
                       + COUT + h * DH + lane * 8;
    float dot = 0.0f;
#pragma unroll
    for (int e = 0; e < 8; ++e) dot += (float)qp[e] * (float)kp[e];
#pragma unroll
    for (int off = 16; off > 0; off >>= 1) dot += __shfl_xor(dot, off, 32);
    s[j] = dot + rpb[h * (2 * KW - 1) + (start + j - i + (KW - 1))];
  }

  float mx = s[0];
#pragma unroll
  for (int j = 1; j < KW; ++j) mx = fmaxf(mx, s[j]);
  float w[KW];
  float sum = 0.0f;
#pragma unroll
  for (int j = 0; j < KW; ++j) { w[j] = __expf(s[j] - mx); sum += w[j]; }
  float inv = 1.0f / sum;

  float acc[8] = {0.f, 0.f, 0.f, 0.f, 0.f, 0.f, 0.f, 0.f};
#pragma unroll
  for (int j = 0; j < KW; ++j) {
    const __bf16* vp = qkv + ((size_t)g * NPOS + start + j) * (3 * COUT)
                       + 2 * COUT + h * DH + lane * 8;
    float wj = w[j] * inv;
#pragma unroll
    for (int e = 0; e < 8; ++e) acc[e] += wj * (float)vp[e];
  }

  __bf16* op = attout + row * COUT + h * DH + lane * 8;
#pragma unroll
  for (int e = 0; e < 8; ++e) op[e] = (__bf16)acc[e];
}

// ---------------- proj GEMM: o_relu = relu(att_out @ proj_w + b) ----------------

__global__ __launch_bounds__(cfg::NT) void proj_gemm_k(const __bf16* __restrict__ A,
                                                       const __bf16* __restrict__ Bw,
                                                       const float* __restrict__ bias,
                                                       __bf16* __restrict__ out) {
  using namespace cfg;
  constexpr int ND = COUT;
  constexpr int KD = COUT;
  constexpr int TN = ND / BN;   // 4
  __shared__ __align__(16) __bf16 sA[BM * LDK];
  __shared__ __align__(16) __bf16 sB[BN * LDK];

  int bid = blockIdx.x;
  int tm = bid / TN, tn = bid % TN;
  int m0 = tm * BM, n0 = tn * BN;
  int tid = threadIdx.x, lane = tid & 31, wave = tid >> 5;
  int wm = wave >> 1, wn = wave & 1;

  v8f acc[2][4] = {};

  for (int c0 = 0; c0 < KD; c0 += BK) {
    int rowA = tid >> 1;
    int ha   = (tid & 1) * 16;
    {
      int m = m0 + rowA;
      uint4 d0 = {0u, 0u, 0u, 0u}, d1 = {0u, 0u, 0u, 0u};
      if (m < MATT) {
        const __bf16* src = A + (size_t)m * KD + c0 + ha;
        d0 = *(const uint4*)(src);
        d1 = *(const uint4*)(src + 8);
        if (c0 + BK < KD) __builtin_prefetch(src + BK, 0, 1);
      }
      *(uint4*)(sA + rowA * LDK + ha)     = d0;
      *(uint4*)(sA + rowA * LDK + ha + 8) = d1;
    }
    {
      const __bf16* src = Bw + (size_t)(n0 + rowA) * KD + c0 + ha;
      *(uint4*)(sB + rowA * LDK + ha)     = *(const uint4*)(src);
      *(uint4*)(sB + rowA * LDK + ha + 8) = *(const uint4*)(src + 8);
    }
    __syncthreads();
    v16bf a0 = ldsA_frag(sA, wm * 32);
    v16bf a1 = ldsA_frag(sA, wm * 32 + 16);
    v16bf bf[4];
#pragma unroll
    for (int ni = 0; ni < 4; ++ni) bf[ni] = ldsB_frag(sB, wn * 64 + ni * 16);
#pragma unroll
    for (int ni = 0; ni < 4; ++ni) {
      acc[0][ni] = wmma_bf16(a0, bf[ni], acc[0][ni]);
      acc[1][ni] = wmma_bf16(a1, bf[ni], acc[1][ni]);
    }
    __syncthreads();
  }

  int lr = lane & 15, lhi = (lane >> 4) & 1;
#pragma unroll
  for (int mi = 0; mi < 2; ++mi) {
#pragma unroll
    for (int ni = 0; ni < 4; ++ni) {
      int cg = n0 + wn * 64 + ni * 16 + lr;
      float bv = bias[cg];
#pragma unroll
      for (int i = 0; i < 8; ++i) {
        int m = m0 + wm * 32 + mi * 16 + i + 8 * lhi;
        if (m < MATT) {
          float v = fmaxf(acc[mi][ni][i] + bv, 0.0f);
          out[(size_t)m * ND + cg] = (__bf16)v;
        }
      }
    }
  }
}

// ---------------- final GEMM: out = relu(o_relu + (x @ down_w + down_b)) ----------------

__global__ __launch_bounds__(cfg::NT) void final_gemm_k(const __bf16* __restrict__ xT,
                                                        const __bf16* __restrict__ Bw,
                                                        const float* __restrict__ bias,
                                                        const __bf16* __restrict__ orelu,
                                                        float* __restrict__ out) {
  using namespace cfg;
  constexpr int ND = COUT;
  constexpr int KD = CIN;
  constexpr int TN = ND / BN;   // 4
  __shared__ __align__(16) __bf16 sA[BM * LDK];
  __shared__ __align__(16) __bf16 sB[BN * LDK];

  int bid = blockIdx.x;
  int tm = bid / TN, tn = bid % TN;
  int m0 = tm * BM, n0 = tn * BN;   // m = b*L + l flat rows, exact multiple
  int tid = threadIdx.x, lane = tid & 31, wave = tid >> 5;
  int wm = wave >> 1, wn = wave & 1;

  v8f acc[2][4] = {};

  for (int c0 = 0; c0 < KD; c0 += BK) {
    int rowA = tid >> 1;
    int ha   = (tid & 1) * 16;
    {
      const __bf16* src = xT + (size_t)(m0 + rowA) * KD + c0 + ha;
      *(uint4*)(sA + rowA * LDK + ha)     = *(const uint4*)(src);
      *(uint4*)(sA + rowA * LDK + ha + 8) = *(const uint4*)(src + 8);
      if (c0 + BK < KD) __builtin_prefetch(src + BK, 0, 1);
    }
    {
      const __bf16* src = Bw + (size_t)(n0 + rowA) * KD + c0 + ha;
      *(uint4*)(sB + rowA * LDK + ha)     = *(const uint4*)(src);
      *(uint4*)(sB + rowA * LDK + ha + 8) = *(const uint4*)(src + 8);
    }
    __syncthreads();
    v16bf a0 = ldsA_frag(sA, wm * 32);
    v16bf a1 = ldsA_frag(sA, wm * 32 + 16);
    v16bf bf[4];
#pragma unroll
    for (int ni = 0; ni < 4; ++ni) bf[ni] = ldsB_frag(sB, wn * 64 + ni * 16);
#pragma unroll
    for (int ni = 0; ni < 4; ++ni) {
      acc[0][ni] = wmma_bf16(a0, bf[ni], acc[0][ni]);
      acc[1][ni] = wmma_bf16(a1, bf[ni], acc[1][ni]);
    }
    __syncthreads();
  }

  int lr = lane & 15, lhi = (lane >> 4) & 1;
#pragma unroll
  for (int mi = 0; mi < 2; ++mi) {
#pragma unroll
    for (int ni = 0; ni < 4; ++ni) {
      int cg = n0 + wn * 64 + ni * 16 + lr;      // output channel o
      float bv = bias[cg];
#pragma unroll
      for (int i = 0; i < 8; ++i) {
        int m = m0 + wm * 32 + mi * 16 + i + 8 * lhi;
        int b = m >> 12;         // / L
        int l = m & (L - 1);
        int g = b * DIL + (l & (DIL - 1));
        int ip = l / DIL;
        float ov = (float)orelu[((size_t)g * NPOS + ip) * COUT + cg];
        float res = acc[mi][ni][i] + bv;
        out[((size_t)b * COUT + cg) * L + l] = fmaxf(ov + res, 0.0f);
      }
    }
  }
}

// ---------------- host launcher ----------------

extern "C" void kernel_launch(void* const* d_in, const int* in_sizes, int n_in,
                              void* d_out, int out_size, void* d_ws, size_t ws_size,
                              hipStream_t stream) {
  using namespace cfg;
  const float* x    = (const float*)d_in[0];
  const float* w1   = (const float*)d_in[1];
  const float* b1   = (const float*)d_in[2];
  const float* wq   = (const float*)d_in[3];
  const float* bq   = (const float*)d_in[4];
  const float* rpb  = (const float*)d_in[5];
  const float* wp   = (const float*)d_in[6];
  const float* bp   = (const float*)d_in[7];
  const float* wd   = (const float*)d_in[8];
  const float* bd   = (const float*)d_in[9];
  float* out = (float*)d_out;

  char* ws = (char*)d_ws;
  size_t off = 0;
  __bf16* xT     = (__bf16*)(ws + off); off += (size_t)B * L * CIN * 2;          // 33.6 MB
  __bf16* wconv  = (__bf16*)(ws + off); off += (size_t)KW * COUT * CIN * 2;      // 3.7 MB
  __bf16* wqkv   = (__bf16*)(ws + off); off += (size_t)3 * COUT * COUT * 2;      // 1.6 MB
  __bf16* wproj  = (__bf16*)(ws + off); off += (size_t)COUT * COUT * 2;          // 0.5 MB
  __bf16* wdown  = (__bf16*)(ws + off); off += (size_t)COUT * CIN * 2;           // 0.5 MB
  __bf16* attx   = (__bf16*)(ws + off); off += (size_t)MATT * COUT * 2;          // 33.8 MB
  __bf16* qkvbuf = (__bf16*)(ws + off); off += (size_t)MATT * 3 * COUT * 2;      // 101 MB
  __bf16* attout = (__bf16*)(ws + off); off += (size_t)MATT * COUT * 2;          // 33.8 MB
  __bf16* orelu  = (__bf16*)(ws + off); off += (size_t)MATT * COUT * 2;          // 33.8 MB

  // pack inputs to bf16 / GEMM layouts
  pack_x_k<<<(B * CIN * L + 255) / 256, 256, 0, stream>>>(x, xT);
  pack_wconv_k<<<(COUT * CIN * KW + 255) / 256, 256, 0, stream>>>(w1, wconv);
  pack_bmat_k<<<(COUT * 3 * COUT + 255) / 256, 256, 0, stream>>>(wq, wqkv, COUT, 3 * COUT);
  pack_bmat_k<<<(COUT * COUT + 255) / 256, 256, 0, stream>>>(wp, wproj, COUT, COUT);
  pack_wdown_k<<<(COUT * CIN + 255) / 256, 256, 0, stream>>>(wd, wdown);
  zero_edge_k<<<(B * PAD * COUT + 255) / 256, 256, 0, stream>>>(attx);

  // dilated conv -> relu -> grouped attention layout
  conv_gemm_k<<<B * (L / BM) * (COUT / BN), NT, 0, stream>>>(xT, wconv, b1, attx);

  // qkv projection
  {
    int mt = (MATT + BM - 1) / BM;       // 258
    int nt = (3 * COUT) / BN;            // 12
    qkv_gemm_k<<<mt * nt, NT, 0, stream>>>(attx, wqkv, bq, qkvbuf);
  }

  // neighborhood attention
  na1d_attn_k<<<(G * NPOS * H + 7) / 8, 256, 0, stream>>>(qkvbuf, rpb, attout);

  // output projection (+relu)
  {
    int mt = (MATT + BM - 1) / BM;       // 258
    int nt = COUT / BN;                  // 4
    proj_gemm_k<<<mt * nt, NT, 0, stream>>>(attout, wproj, bp, orelu);
  }

  // residual 1x1 conv + combine + final relu
  final_gemm_k<<<(B * L / BM) * (COUT / BN), NT, 0, stream>>>(xT, wdown, bd, orelu, out);
}
